// CrossAttentionBlock_41068477284396
// MI455X (gfx1250) — matrix operand order repaired and checked by reading, not compiled, not run on hardware
//
#include <hip/hip_runtime.h>
#include <hip/hip_bf16.h>

// ---------------------------------------------------------------------------
// CrossAttentionBlock for MI455X (gfx1250), bf16 WMMA pipeline.
//   B=8, X=Y=64 (XY=4096), C=128, CTX=768, N=256, H=8, D=64, HID=512
// GEMMs use 64x64 register-blocked tiles (16 wmma / 8 frag loads per K-step).
// ---------------------------------------------------------------------------

typedef __attribute__((ext_vector_type(16))) __bf16 v16bf;
typedef __attribute__((ext_vector_type(8)))  float  v8f;

#define BB    8
#define XY    4096
#define CC    128
#define CTXD  768
#define NCTX  256
#define NH    8
#define HD    64
#define HIDD  512
#define KPAD  288   // 1 null key + 256 ctx keys + 31 zero pad -> 9 * 32 K-steps

__device__ __forceinline__ v8f wmma_bf16(v16bf a, v16bf b, v8f c) {
  // 8 args: (neg_a, A, neg_b, B, c_mod, C, reuse_a, reuse_b)
  return __builtin_amdgcn_wmma_f32_16x16x32_bf16(false, a, false, b, (short)0, c,
                                                 false, false);
}

// A fragment: 16x32 tile of a row-major [*, ld] bf16 matrix; base -> (0,0).
// k order used here (same in frag_bT): lanes 0-15 hold K 0-15, lanes 16-31 K 16-31,
// row M = lane%16.
__device__ __forceinline__ v16bf frag_a(const __bf16* base, int ld) {
  int lane = threadIdx.x & 31, half = lane >> 4, m = lane & 15;
  const __bf16* p = base + (size_t)m * ld + half * 16;
  v16bf r;
#pragma unroll
  for (int e = 0; e < 16; ++e) r[e] = p[e];
  return r;
}

// B fragment loaded from a B^T layout [N][K] (ld = K stride); base -> (k=0,n=0).
// k order used here (same as frag_a): lanes 0-15 K=0..15, lanes 16-31 K=16..31, col N = lane%16.
__device__ __forceinline__ v16bf frag_bT(const __bf16* base, int ld) {
  int lane = threadIdx.x & 31, half = lane >> 4, n = lane & 15;
  const __bf16* p = base + (size_t)n * ld + half * 16;
  v16bf r;
#pragma unroll
  for (int e = 0; e < 16; ++e) r[e] = p[e];
  return r;
}

// 64x64 register-blocked GEMM core: acc[i][j] += A[64xK] * B^T[64xK]^T
__device__ __forceinline__ void gemm_block64(const __bf16* __restrict__ Ab, int lda,
                                             const __bf16* __restrict__ Bb, int ldb,
                                             int K, v8f acc[4][4]) {
  for (int k0 = 0; k0 < K; k0 += 32) {
    __builtin_prefetch(Ab + k0 + 64, 0, 1);   // global_prefetch_b8
    v16bf a[4], b[4];
#pragma unroll
    for (int i = 0; i < 4; ++i) a[i] = frag_a(Ab + (size_t)(i * 16) * lda + k0, lda);
#pragma unroll
    for (int j = 0; j < 4; ++j) b[j] = frag_bT(Bb + (size_t)(j * 16) * ldb + k0, ldb);
#pragma unroll
    for (int i = 0; i < 4; ++i)
#pragma unroll
      for (int j = 0; j < 4; ++j)
        acc[i][j] = wmma_bf16(a[i], b[j], acc[i][j]);
  }
}

// ------------------------- weight transpose fp32 -> bf16 B^T ----------------
__global__ __launch_bounds__(256) void k_transpose_bf16(
    const float* __restrict__ in, __bf16* __restrict__ out, int R, int Ccols) {
  int idx = blockIdx.x * 256 + threadIdx.x;
  if (idx < R * Ccols) {
    int r = idx / Ccols, c = idx - r * Ccols;
    out[(size_t)c * R + r] = (__bf16)in[idx];
  }
}

// ------------------------- LayerNorm over C=128 (one wave per row) ----------
__global__ __launch_bounds__(256) void k_ln_x(
    const float* __restrict__ xt, const float* __restrict__ w,
    const float* __restrict__ b, float* __restrict__ xnf,
    __bf16* __restrict__ xnb) {
  int row  = blockIdx.x * 8 + (threadIdx.x >> 5);
  int lane = threadIdx.x & 31;
  const float4 v = ((const float4*)(xt + (size_t)row * CC))[lane];
  float s  = v.x + v.y + v.z + v.w;
  float ss = v.x * v.x + v.y * v.y + v.z * v.z + v.w * v.w;
#pragma unroll
  for (int o = 16; o; o >>= 1) { s += __shfl_xor(s, o, 32); ss += __shfl_xor(ss, o, 32); }
  float mu = s * (1.0f / CC);
  float rs = rsqrtf(ss * (1.0f / CC) - mu * mu + 1e-5f);
  float e4[4] = {v.x, v.y, v.z, v.w};
#pragma unroll
  for (int i = 0; i < 4; ++i) {
    int   c = lane * 4 + i;
    float o = (e4[i] - mu) * rs * w[c] + b[c];
    xnf[(size_t)row * CC + c] = o;
    xnb[(size_t)row * CC + c] = (__bf16)o;
  }
}

// ------------------------- LayerNorm over CTX=768 (one wave per row) --------
__global__ __launch_bounds__(256) void k_ln_ctx(
    const float* __restrict__ ctx, const float* __restrict__ w,
    const float* __restrict__ b, __bf16* __restrict__ cnb) {
  int row  = blockIdx.x * 8 + (threadIdx.x >> 5);
  int lane = threadIdx.x & 31;
  const float* p = ctx + (size_t)row * CTXD;
  float vals[24];
  float s = 0.f, ss = 0.f;
#pragma unroll
  for (int i = 0; i < 24; ++i) {
    float x = p[lane + i * 32];
    vals[i] = x; s += x; ss += x * x;
  }
#pragma unroll
  for (int o = 16; o; o >>= 1) { s += __shfl_xor(s, o, 32); ss += __shfl_xor(ss, o, 32); }
  float mu = s * (1.0f / CTXD);
  float rs = rsqrtf(ss * (1.0f / CTXD) - mu * mu + 1e-5f);
#pragma unroll
  for (int i = 0; i < 24; ++i) {
    int c = lane + i * 32;
    cnb[(size_t)row * CTXD + c] = (__bf16)((vals[i] - mu) * rs * w[c] + b[c]);
  }
}

// ------------------------- init null-KV + zero pad --------------------------
__global__ __launch_bounds__(256) void k_init_kv(
    const float* __restrict__ null_kv, __bf16* __restrict__ Kb,
    __bf16* __restrict__ Vt) {
  int idx = blockIdx.x * 256 + threadIdx.x;   // 64 bh * 64 d = 4096
  if (idx >= BB * NH * HD) return;
  int bh = idx >> 6, d = idx & 63;
  Kb[((size_t)bh * KPAD + 0) * HD + d] = (__bf16)null_kv[d];        // null_kv[0][d]
  Vt[((size_t)bh * HD + d) * KPAD + 0] = (__bf16)null_kv[HD + d];   // null_kv[1][d]
  for (int j = NCTX + 1; j < KPAD; ++j) {
    Kb[((size_t)bh * KPAD + j) * HD + d] = (__bf16)0.f;
    Vt[((size_t)bh * HD + d) * KPAD + j] = (__bf16)0.f;
  }
}

// ------------------------- Q = xn @ Wq  (32768x512, K=128) ------------------
// 64x64 blocks: 512 x 8 blocks, 1 wave each, 4 waves / 128-thread workgroup.
__global__ __launch_bounds__(128, 1) void k_gemm_q(
    const __bf16* __restrict__ A, const __bf16* __restrict__ BT,
    __bf16* __restrict__ C) {
  int tile = blockIdx.x * 4 + (threadIdx.x >> 5);
  int tn = tile & 7, tm = tile >> 3;
  v8f acc[4][4] = {};
  gemm_block64(A + (size_t)(tm * 64) * CC, CC, BT + (size_t)(tn * 64) * CC, CC,
               CC, acc);
  int lane = threadIdx.x & 31, half = lane >> 4, nn = lane & 15;
#pragma unroll
  for (int i = 0; i < 4; ++i)
#pragma unroll
    for (int j = 0; j < 4; ++j)
#pragma unroll
      for (int r = 0; r < 8; ++r)
        C[(size_t)(tm * 64 + i * 16 + half * 8 + r) * HIDD + tn * 64 + j * 16 + nn] =
            (__bf16)acc[i][j][r];
}

// -------- KV = cn @ Wkv (2048x1024, K=768) -> scatter into Kb / Vt ----------
__global__ __launch_bounds__(128, 1) void k_gemm_kv(
    const __bf16* __restrict__ A, const __bf16* __restrict__ BT,
    __bf16* __restrict__ Kb, __bf16* __restrict__ Vt) {
  int tile = blockIdx.x * 4 + (threadIdx.x >> 5);   // 32 x 16 blocks
  int tn = tile & 15, tm = tile >> 4;
  v8f acc[4][4] = {};
  gemm_block64(A + (size_t)(tm * 64) * CTXD, CTXD, BT + (size_t)(tn * 64) * CTXD,
               CTXD, CTXD, acc);
  int lane = threadIdx.x & 31, half = lane >> 4, nn = lane & 15;
#pragma unroll
  for (int i = 0; i < 4; ++i)
#pragma unroll
    for (int j = 0; j < 4; ++j) {
      int col = tn * 64 + j * 16 + nn;
#pragma unroll
      for (int r = 0; r < 8; ++r) {
        int row = tm * 64 + i * 16 + half * 8 + r;   // row = b*256 + n
        int b = row >> 8, n = row & 255, jj = n + 1;
        __bf16 v = (__bf16)acc[i][j][r];
        if (col < HIDD) {                            // K half
          int h = col >> 6, d = col & 63;
          Kb[(((size_t)(b * NH + h)) * KPAD + jj) * HD + d] = v;
        } else {                                     // V half (store transposed)
          int c2 = col - HIDD, h = c2 >> 6, d = c2 & 63;
          Vt[(((size_t)(b * NH + h)) * HD + d) * KPAD + jj] = v;
        }
      }
    }
}

// ------------------------- attention: one wave per (b,h,16-row block) -------
__global__ __launch_bounds__(32) void k_attn(
    const __bf16* __restrict__ Qb, const __bf16* __restrict__ Kb,
    const __bf16* __restrict__ Vt, const unsigned char* __restrict__ mask,
    __bf16* __restrict__ Obf) {
  __shared__ float  S[16][KPAD];
  __shared__ __bf16 P[16][KPAD];
  int tile = blockIdx.x;                     // b*2048 + h*256 + rb
  int rb = tile & 255, h = (tile >> 8) & 7, b = tile >> 11;
  int row0 = b * XY + rb * 16;               // rows in Qb
  const __bf16* Abase = Qb + (size_t)row0 * HIDD + h * HD;
  const __bf16* Kbh = Kb + (size_t)(b * NH + h) * KPAD * HD;   // [j][d] == B^T
  const __bf16* Vth = Vt + (size_t)(b * NH + h) * HD * KPAD;   // [d][j] == B^T
  int lane = threadIdx.x, half = lane >> 4, nn = lane & 15;
  const float scale = 0.125f;                // D^-0.5

  // Q fragments are invariant over key tiles: load once (16x64 = 2 frags).
  v16bf qa[2];
#pragma unroll
  for (int k = 0; k < 2; ++k) qa[k] = frag_a(Abase + k * 32, HIDD);

  // S = (Q @ K^T) * scale, masked
  for (int jt = 0; jt < KPAD / 16; ++jt) {
    v8f acc = {};
#pragma unroll
    for (int k = 0; k < 2; ++k) {
      v16bf bb = frag_bT(Kbh + (size_t)(jt * 16) * HD + k * 32, HD);
      acc = wmma_bf16(qa[k], bb, acc);
    }
    int j = jt * 16 + nn;
    bool valid = (j == 0) || (j <= NCTX && mask[b * NCTX + (j - 1)] != 0);
#pragma unroll
    for (int r = 0; r < 8; ++r)
      S[half * 8 + r][j] = valid ? acc[r] * scale : -3.0e38f;
  }
  __syncthreads();

  // softmax over j (lane pair r / r+16 splits the 288 columns)
  {
    int r = lane & 15, hh = lane >> 4;
    int c0 = hh * (KPAD / 2);
    float mx = -3.4e38f;
    for (int c = 0; c < KPAD / 2; ++c) mx = fmaxf(mx, S[r][c0 + c]);
    mx = fmaxf(mx, __shfl_xor(mx, 16, 32));
    float sum = 0.f;
    for (int c = 0; c < KPAD / 2; ++c) {
      float e = __expf(S[r][c0 + c] - mx);
      S[r][c0 + c] = e; sum += e;
    }
    sum += __shfl_xor(sum, 16, 32);
    float inv = 1.0f / sum;
    for (int c = 0; c < KPAD / 2; ++c) P[r][c0 + c] = (__bf16)(S[r][c0 + c] * inv);
  }
  __syncthreads();

  // O = P @ V  (K = 288): one LDS A-frag feeds all 4 D-column tiles.
  v8f oacc[4] = {};
  for (int k0 = 0; k0 < KPAD; k0 += 32) {
    v16bf a = frag_a(&P[0][0] + k0, KPAD);
#pragma unroll
    for (int nt = 0; nt < 4; ++nt) {
      v16bf bb = frag_bT(Vth + (size_t)(nt * 16) * KPAD + k0, KPAD);
      oacc[nt] = wmma_bf16(a, bb, oacc[nt]);
    }
  }
#pragma unroll
  for (int nt = 0; nt < 4; ++nt)
#pragma unroll
    for (int r = 0; r < 8; ++r) {
      int row = row0 + half * 8 + r;
      Obf[(size_t)row * HIDD + h * HD + nt * 16 + nn] = (__bf16)oacc[nt][r];
    }
}

// -------- final: out = O @ Wout + bout + xn  (32768x128, K=512, fp32 out) ---
__global__ __launch_bounds__(128, 1) void k_gemm_out(
    const __bf16* __restrict__ A, const __bf16* __restrict__ BT,
    const float* __restrict__ bout, const float* __restrict__ xnf,
    float* __restrict__ out) {
  int tile = blockIdx.x * 4 + (threadIdx.x >> 5);   // 512 x 2 blocks
  int tn = tile & 1, tm = tile >> 1;
  v8f acc[4][4] = {};
  gemm_block64(A + (size_t)(tm * 64) * HIDD, HIDD, BT + (size_t)(tn * 64) * HIDD,
               HIDD, HIDD, acc);
  int lane = threadIdx.x & 31, half = lane >> 4, nn = lane & 15;
#pragma unroll
  for (int i = 0; i < 4; ++i)
#pragma unroll
    for (int j = 0; j < 4; ++j) {
      int col = tn * 64 + j * 16 + nn;
#pragma unroll
      for (int r = 0; r < 8; ++r) {
        int row = tm * 64 + i * 16 + half * 8 + r;
        out[(size_t)row * CC + col] =
            acc[i][j][r] + bout[col] + xnf[(size_t)row * CC + col];
      }
    }
}

// ---------------------------------------------------------------------------
extern "C" void kernel_launch(void* const* d_in, const int* in_sizes, int n_in,
                              void* d_out, int out_size, void* d_ws, size_t ws_size,
                              hipStream_t stream) {
  // inputs (setup_inputs order)
  const float* xt      = (const float*)d_in[0];
  const float* context = (const float*)d_in[1];
  const unsigned char* mask = (const unsigned char*)d_in[2]; // jnp bool = 1 byte
  const float* norm_w  = (const float*)d_in[3];
  const float* norm_b  = (const float*)d_in[4];
  const float* cnorm_w = (const float*)d_in[5];
  const float* cnorm_b = (const float*)d_in[6];
  const float* Wq      = (const float*)d_in[7];
  const float* Wkv     = (const float*)d_in[8];
  const float* null_kv = (const float*)d_in[9];
  const float* Wout    = (const float*)d_in[10];
  const float* bout    = (const float*)d_in[11];
  float* out = (float*)d_out;

  // workspace carve-out (256B aligned)
  uintptr_t p = (uintptr_t)d_ws;
  auto carve = [&](size_t bytes) -> void* {
    void* r = (void*)p;
    p += (bytes + 255) & ~(size_t)255;
    return r;
  };
  const size_t ROWS = (size_t)BB * XY;                 // 32768
  float*  xnf   = (float*) carve(ROWS * CC * 4);
  __bf16* xnb   = (__bf16*)carve(ROWS * CC * 2);
  __bf16* cnb   = (__bf16*)carve((size_t)BB * NCTX * CTXD * 2);
  __bf16* WqT   = (__bf16*)carve((size_t)HIDD * CC * 2);
  __bf16* WkvT  = (__bf16*)carve((size_t)2 * HIDD * CTXD * 2);
  __bf16* WoutT = (__bf16*)carve((size_t)CC * HIDD * 2);
  __bf16* Qb    = (__bf16*)carve(ROWS * HIDD * 2);
  __bf16* Kb    = (__bf16*)carve((size_t)BB * NH * KPAD * HD * 2);
  __bf16* Vt    = (__bf16*)carve((size_t)BB * NH * HD * KPAD * 2);
  __bf16* Obf   = (__bf16*)carve(ROWS * HIDD * 2);
  (void)ws_size; (void)n_in; (void)in_sizes; (void)out_size;

  // weight transposes -> bf16 B^T
  k_transpose_bf16<<<(CC * HIDD + 255) / 256, 256, 0, stream>>>(Wq, WqT, CC, HIDD);
  k_transpose_bf16<<<(CTXD * 2 * HIDD + 255) / 256, 256, 0, stream>>>(Wkv, WkvT, CTXD, 2 * HIDD);
  k_transpose_bf16<<<(HIDD * CC + 255) / 256, 256, 0, stream>>>(Wout, WoutT, HIDD, CC);

  // layernorms
  k_ln_x<<<(int)(ROWS / 8), 256, 0, stream>>>(xt, norm_w, norm_b, xnf, xnb);
  k_ln_ctx<<<(BB * NCTX) / 8, 256, 0, stream>>>(context, cnorm_w, cnorm_b, cnb);

  // null kv + zero padding of key dim
  k_init_kv<<<(BB * NH * HD + 255) / 256, 256, 0, stream>>>(null_kv, Kb, Vt);

  // projections: 64x64 blocks, 4 waves per 128-thread workgroup
  k_gemm_q<<<(int)((ROWS / 64) * (HIDD / 64) / 4), 128, 0, stream>>>(xnb, WqT, Qb);
  k_gemm_kv<<<(BB * NCTX / 64) * (2 * HIDD / 64) / 4, 128, 0, stream>>>(cnb, WkvT, Kb, Vt);

  // attention: one wave per (b, h, 16-query block)
  k_attn<<<BB * NH * (XY / 16), 32, 0, stream>>>(Qb, Kb, Vt, mask, Obf);

  // output projection + bias + residual
  k_gemm_out<<<(int)((ROWS / 64) * (CC / 64) / 4), 128, 0, stream>>>(Obf, WoutT, bout, xnf, out);
}
